// PointConv_42219528520132
// MI455X (gfx1250) — matrix operand order repaired and checked by reading, not compiled, not run on hardware
//
#include <hip/hip_runtime.h>
#include <hip/hip_bf16.h>

// ---------------- problem constants ----------------
#define BB   4
#define NN   8192
#define SS   2048
#define KK   32
#define DD   61
#define EPSF 1e-5f

// d_out layout (floats): new_xyz [B,3,S] | out [B,128,S] | grouped_xyz [B,S,K,3] | sample_idx [B,S]
#define OFF_NEWXYZ  0
#define OFF_OUT     (BB*3*SS)                 // 24576
#define OFF_GROUP   (OFF_OUT + BB*128*SS)     // 1073152
#define OFF_SIDX    (OFF_GROUP + BB*SS*KK*3)  // 1859584

typedef __attribute__((ext_vector_type(2)))  float  v2f;
typedef __attribute__((ext_vector_type(8)))  float  v8f;
typedef __attribute__((ext_vector_type(16))) __bf16 v16bf;

// ---------------- helpers ----------------
__device__ __forceinline__ unsigned int ford(float f) {
    unsigned int u = __float_as_uint(f);
    return (u & 0x80000000u) ? ~u : (u | 0x80000000u);
}
__device__ __forceinline__ unsigned short f2bfbits(float f) {
    unsigned int u = __float_as_uint(f);
    unsigned int r = u + 0x7FFFu + ((u >> 16) & 1u);   // round-to-nearest-even
    return (unsigned short)(r >> 16);
}
__device__ __forceinline__ float bnrelu(float acc, float bo, float go, float beo) {
    float y = (acc + bo) * (go * rsqrtf(1.f + EPSF)) + beo;
    return fmaxf(y, 0.f);
}

// 16 bf16 halves assembled from two 16-byte vector loads
union Frag16 {
    uint4 q[2];
    v16bf v;
};

// ---------------- K1: gather sampled centers ----------------
__global__ void centers_kernel(const float* __restrict__ xyz, const int* __restrict__ sidx,
                               float* __restrict__ nxyz_out, float* __restrict__ nxyz_ws,
                               int* __restrict__ sidx_out) {
    int i = blockIdx.x * 256 + threadIdx.x;
    if (i >= BB * SS) return;
    int b = i >> 11, s = i & (SS - 1);
    int n = sidx[i];
    const float* xb = xyz + (size_t)b * 3 * NN;
    float x = xb[n], y = xb[NN + n], z = xb[2 * NN + n];
    nxyz_out[((size_t)b * 3 + 0) * SS + s] = x;
    nxyz_out[((size_t)b * 3 + 1) * SS + s] = y;
    nxyz_out[((size_t)b * 3 + 2) * SS + s] = z;
    nxyz_ws[i * 3 + 0] = x; nxyz_ws[i * 3 + 1] = y; nxyz_ws[i * 3 + 2] = z;
    sidx_out[i] = n;
}

// ---------------- K2: kNN (top-32 smallest squared distance) ----------------
__global__ void __launch_bounds__(256) knn_kernel(const float* __restrict__ xyz,
                                                  const float* __restrict__ nxyz_ws,
                                                  int* __restrict__ idx_ws,
                                                  float* __restrict__ grouped) {
    __shared__ unsigned int       keys[NN];     // orderable distance bits
    __shared__ unsigned long long part[256];    // per-thread packed (key,idx) min
    __shared__ unsigned long long red[256];
    __shared__ int                sel[KK];
    int tid = threadIdx.x;
    int g = blockIdx.x;               // b*S + s
    int b = g >> 11;
    const float* xb = xyz + (size_t)b * 3 * NN;
    float cx = nxyz_ws[g * 3 + 0], cy = nxyz_ws[g * 3 + 1], cz = nxyz_ws[g * 3 + 2];
    float cc = cx * cx + cy * cy + cz * cz;

    unsigned long long mymin = ~0ull;
    for (int j = 0; j < NN / 256; ++j) {
        int n = (j << 8) + tid;       // owner thread == n & 255
        float px = xb[n], py = xb[NN + n], pz = xb[2 * NN + n];
        float d = -2.f * (cx * px + cy * py + cz * pz) + cc + (px * px + py * py + pz * pz);
        unsigned int k32 = ford(d);
        keys[n] = k32;
        unsigned long long pk = ((unsigned long long)k32 << 32) | (unsigned int)n;
        if (pk < mymin) mymin = pk;
    }
    part[tid] = mymin;
    __syncthreads();

    for (int r = 0; r < KK; ++r) {
        red[tid] = part[tid];
        __syncthreads();
        for (int st = 128; st > 0; st >>= 1) {
            if (tid < st) { unsigned long long o = red[tid + st]; if (o < red[tid]) red[tid] = o; }
            __syncthreads();
        }
        unsigned long long w = red[0];
        int n = (int)(w & 0xFFFFFFFFu);
        if (tid == 0) sel[r] = n;
        __syncthreads();
        if (tid == (n & 255)) {       // winner rescans only its own strided slice
            keys[n] = 0xFFFFFFFFu;
            unsigned long long m = ~0ull;
            for (int j = 0; j < NN / 256; ++j) {
                int nn = (j << 8) + tid;
                unsigned long long pk = ((unsigned long long)keys[nn] << 32) | (unsigned int)nn;
                if (pk < m) m = pk;
            }
            part[tid] = m;
        }
        __syncthreads();
    }

    if (tid < KK) {
        int n = sel[tid];
        idx_ws[(size_t)g * KK + tid] = n;
        size_t base = ((size_t)g * KK + tid) * 3;
        grouped[base + 0] = xb[n];
        grouped[base + 1] = xb[NN + n];
        grouped[base + 2] = xb[2 * NN + n];
    }
}

// ---------------- K4: lin_w [128,2048] f32 -> bf16, fragment-swizzled [kblk=64][o=128][kk=32] ----------
// Lane (hi,row16) of output-tile `otile` reads B halves K = kblk*32 + hi*16 + 0..15 at fixed o:
// contiguous 32 bytes in this layout -> two global_load_b128 per fragment.
__global__ void convert_linw_kernel(const float* __restrict__ lin_w, unsigned short* __restrict__ lin_sw) {
    int i = blockIdx.x * 256 + threadIdx.x;
    if (i >= 64 * 128 * 32) return;
    int t   = i >> 12;          // k-block
    int rem = i & 4095;
    int o   = rem >> 5;
    int kk  = rem & 31;
    lin_sw[i] = f2bfbits(lin_w[(size_t)o * 2048 + t * 32 + kk]);
}

// ---------------- fp32 WMMA MLP layer: dst = relu(BN(src @ W^T + b)) ----------------
__device__ __forceinline__ void mlp_layer(const float* src, float* dst, int cin, int cout,
                                          const float* __restrict__ W, const float* __restrict__ bb,
                                          const float* __restrict__ gg, const float* __restrict__ be,
                                          int wave, int row16, int hi) {
    int ntiles = (cout >> 4) * 2;                     // 2 M-tiles (K=32 rows) x cout/16 N-tiles
    for (int t = wave; t < ntiles; t += 4) {          // wave-uniform: EXEC stays all-ones
        int m0 = (t & 1) << 4;
        int n0 = (t >> 1) << 4;
        v8f acc = {};
        for (int kc = 0; kc < cin; kc += 4) {
            int ka = kc + (hi << 1);                  // A: lanes16-31 hold K+2,K+3
            v2f av, bv;
            av.x = src[(m0 + row16) * 128 + ka];
            av.y = src[(m0 + row16) * 128 + ka + 1];
            bv.x = W[(n0 + row16) * cin + ka];        // B[c][o] = W[o][c]
            bv.y = W[(n0 + row16) * cin + ka + 1];
            acc = __builtin_amdgcn_wmma_f32_16x16x4_f32(false, av, false, bv, (short)0, acc, false, false);
        }
        int o = n0 + row16;
        float bo = bb[o], go = gg[o], beo = be[o];
#pragma unroll
        for (int r = 0; r < 8; ++r) {
            dst[(m0 + r + (hi << 3)) * 128 + o] = bnrelu(acc[r], bo, go, beo);
        }
    }
}

// ---------------- K3: per-group WeightNet + feature MLP + aggregation ----------------
__global__ void __launch_bounds__(128) group_kernel(
    const float* __restrict__ points,
    const float* __restrict__ grouped, const float* __restrict__ nxyz_ws, const int* __restrict__ idx_ws,
    const float* w0w, const float* w0b, const float* w0g, const float* w0e,
    const float* w1w, const float* w1b, const float* w1g, const float* w1e,
    const float* w2w, const float* w2b, const float* w2g, const float* w2e,
    const float* m0w, const float* m0b, const float* m0g, const float* m0e,
    const float* m1w, const float* m1b, const float* m1g, const float* m1e,
    const float* m2w, const float* m2b, const float* m2g, const float* m2e,
    unsigned short* __restrict__ agg_bf) {
    __shared__ float bufA[32 * 128];
    __shared__ float bufB[32 * 128];
    __shared__ float wlds[32 * 16];
    int tid = threadIdx.x;
    int wave = tid >> 5, lane = tid & 31, row16 = lane & 15, hi = lane >> 4;
    int g = blockIdx.x;
    int b = g >> 11;

    // build feats0 [32, 64] = [gnorm(3) | gathered points(61)]
    for (int i = tid; i < 32 * 64; i += 128) {
        int k = i >> 6, c = i & 63;
        int n = idx_ws[(size_t)g * KK + k];
        float v;
        if (c < 3) v = grouped[((size_t)g * KK + k) * 3 + c] - nxyz_ws[g * 3 + c];
        else       v = points[(size_t)b * DD * NN + (size_t)(c - 3) * NN + n];
        bufA[k * 128 + c] = v;
    }
    __syncthreads();

    // WeightNet 3->8->8->16 (tiny; scalar per point)
    if (tid < 32) {
        int k = tid;
        float x0 = bufA[k * 128 + 0], x1 = bufA[k * 128 + 1], x2 = bufA[k * 128 + 2];
        float h0[8], h1[8];
        for (int o = 0; o < 8; ++o) {
            float a = w0w[o * 3] * x0 + w0w[o * 3 + 1] * x1 + w0w[o * 3 + 2] * x2;
            h0[o] = bnrelu(a, w0b[o], w0g[o], w0e[o]);
        }
        for (int o = 0; o < 8; ++o) {
            float a = 0.f;
            for (int c = 0; c < 8; ++c) a += h0[c] * w1w[o * 8 + c];
            h1[o] = bnrelu(a, w1b[o], w1g[o], w1e[o]);
        }
        for (int o = 0; o < 16; ++o) {
            float a = 0.f;
            for (int c = 0; c < 8; ++c) a += h1[c] * w2w[o * 8 + c];
            wlds[k * 16 + o] = bnrelu(a, w2b[o], w2g[o], w2e[o]);
        }
    }

    // feature MLP 64->64->64->128 on fp32 WMMA
    mlp_layer(bufA, bufB, 64, 64, m0w, m0b, m0g, m0e, wave, row16, hi);
    __syncthreads();
    mlp_layer(bufB, bufA, 64, 64, m1w, m1b, m1g, m1e, wave, row16, hi);
    __syncthreads();
    mlp_layer(bufA, bufB, 64, 128, m2w, m2b, m2g, m2e, wave, row16, hi);
    __syncthreads();

    // aggregation: agg[c,w] = sum_k feats3[k][c] * wnet[k][w]  (M=128 -> 8 tiles, N=16, K=32)
    for (int t = wave; t < 8; t += 4) {
        int m0_ = t << 4;
        v8f acc = {};
        for (int kc = 0; kc < 32; kc += 4) {
            int ka = kc + (hi << 1);
            v2f av, bv;
            av.x = bufB[ka * 128 + m0_ + row16];       // A[m][kk] = feats3[kk][m]
            av.y = bufB[(ka + 1) * 128 + m0_ + row16];
            bv.x = wlds[ka * 16 + row16];
            bv.y = wlds[(ka + 1) * 16 + row16];
            acc = __builtin_amdgcn_wmma_f32_16x16x4_f32(false, av, false, bv, (short)0, acc, false, false);
        }
#pragma unroll
        for (int r = 0; r < 8; ++r) {
            int c = m0_ + r + (hi << 3);
            agg_bf[(size_t)g * 2048 + c * 16 + row16] = f2bfbits(acc[r]);  // reshape [C,W]->C*16+w
        }
    }
}

// ---------------- K5: final linear [16 queries,2048] @ bf16 [2048,128], BN+ReLU, transpose ----------------
__global__ void __launch_bounds__(256) final_linear_kernel(
    const unsigned short* __restrict__ agg_bf, const unsigned short* __restrict__ lin_sw,
    const float* __restrict__ lin_b, const float* __restrict__ lin_g, const float* __restrict__ lin_be,
    float* __restrict__ out) {
    int tid = threadIdx.x;
    int wave = tid >> 5, lane = tid & 31, row16 = lane & 15, hi = lane >> 4;
    int qbase = blockIdx.x << 4;        // 16 queries per block
    int otile = wave << 4;              // 8 waves x 16 outputs = 128
    int o = otile + row16;
    const unsigned short* arow = agg_bf + (size_t)(qbase + row16) * 2048;
    const unsigned short* brow = lin_sw + ((size_t)o << 5) + (hi << 4);  // + kblk*4096

    v8f acc = {};
    for (int t = 0; t < 64; ++t) {      // K = 2048 = 64 x 32
        int k0 = t << 5;
        Frag16 a, bf;
        // A 16-bit layout: per lane 16 contiguous halves: K = k0 + hi*16 + 0..15 (matches B)
        a.q[0] = *(const uint4*)(arow + k0 + (hi << 4));
        a.q[1] = *(const uint4*)(arow + k0 + (hi << 4) + 8);
        // B swizzled: 16 contiguous halves per lane: K = k0 + hi*16 + 0..15 at fixed o
        b_load:
        bf.q[0] = *(const uint4*)(brow + ((size_t)t << 12));
        bf.q[1] = *(const uint4*)(brow + ((size_t)t << 12) + 8);
        if (t + 8 < 64) {
            __builtin_prefetch(arow + k0 + 256, 0, 0);            // global_prefetch_b8
        }
        acc = __builtin_amdgcn_wmma_f32_16x16x32_bf16(false, a.v, false, bf.v, (short)0, acc, false, false);
    }
    float bo = lin_b[o], go = lin_g[o], beo = lin_be[o];
#pragma unroll
    for (int r = 0; r < 8; ++r) {
        int q = qbase + r + (hi << 3);
        int b = q >> 11, s = q & (SS - 1);
        out[((size_t)b * 128 + o) * SS + s] = bnrelu(acc[r], bo, go, beo);
    }
}

// ---------------- launch ----------------
extern "C" void kernel_launch(void* const* d_in, const int* in_sizes, int n_in,
                              void* d_out, int out_size, void* d_ws, size_t ws_size,
                              hipStream_t stream) {
    (void)in_sizes; (void)n_in; (void)out_size; (void)ws_size;
    const float* xyz    = (const float*)d_in[0];
    const float* points = (const float*)d_in[1];
    const int*   sidx   = (const int*)d_in[2];
    const float* wn_w[3] = {(const float*)d_in[3],  (const float*)d_in[4],  (const float*)d_in[5]};
    const float* wn_b[3] = {(const float*)d_in[6],  (const float*)d_in[7],  (const float*)d_in[8]};
    const float* wn_g[3] = {(const float*)d_in[9],  (const float*)d_in[10], (const float*)d_in[11]};
    const float* wn_e[3] = {(const float*)d_in[12], (const float*)d_in[13], (const float*)d_in[14]};
    const float* ml_w[3] = {(const float*)d_in[15], (const float*)d_in[16], (const float*)d_in[17]};
    const float* ml_b[3] = {(const float*)d_in[18], (const float*)d_in[19], (const float*)d_in[20]};
    const float* ml_g[3] = {(const float*)d_in[21], (const float*)d_in[22], (const float*)d_in[23]};
    const float* ml_e[3] = {(const float*)d_in[24], (const float*)d_in[25], (const float*)d_in[26]};
    const float* lin_w  = (const float*)d_in[27];
    const float* lin_b  = (const float*)d_in[28];
    const float* lin_g  = (const float*)d_in[29];
    const float* lin_be = (const float*)d_in[30];

    float* out = (float*)d_out;
    char* ws = (char*)d_ws;
    float*          nxyz_ws = (float*)(ws);                       // B*S*3 f32      (98304 B)
    int*            idx_ws  = (int*)(ws + 98304);                 // B*S*K i32      (1048576 B)
    unsigned short* lin_sw  = (unsigned short*)(ws + 1146880);    // 64*128*32 bf16 (524288 B)
    unsigned short* agg_bf  = (unsigned short*)(ws + 1671168);    // B*S*2048 bf16  (33554432 B)

    centers_kernel<<<(BB * SS + 255) / 256, 256, 0, stream>>>(xyz, sidx, out + OFF_NEWXYZ,
                                                              nxyz_ws, (int*)(out + OFF_SIDX));
    knn_kernel<<<BB * SS, 256, 0, stream>>>(xyz, nxyz_ws, idx_ws, out + OFF_GROUP);
    convert_linw_kernel<<<(64 * 128 * 32 + 255) / 256, 256, 0, stream>>>(lin_w, lin_sw);
    group_kernel<<<BB * SS, 128, 0, stream>>>(points, out + OFF_GROUP, nxyz_ws, idx_ws,
        wn_w[0], wn_b[0], wn_g[0], wn_e[0],
        wn_w[1], wn_b[1], wn_g[1], wn_e[1],
        wn_w[2], wn_b[2], wn_g[2], wn_e[2],
        ml_w[0], ml_b[0], ml_g[0], ml_e[0],
        ml_w[1], ml_b[1], ml_g[1], ml_e[1],
        ml_w[2], ml_b[2], ml_g[2], ml_e[2],
        agg_bf);
    final_linear_kernel<<<BB * SS / 16, 256, 0, stream>>>(agg_bf, lin_sw, lin_b, lin_g, lin_be,
                                                          out + OFF_OUT);
}
